// GraphConvolutionSparse_2_24644522344646
// MI455X (gfx1250) — matrix operand order repaired and checked by reading, not compiled, not run on hardware
//
#include <hip/hip_runtime.h>
#include <hip/hip_bf16.h>

// ---------------------------------------------------------------------------
// GraphConvolutionSparse for MI455X (gfx1250, wave32, WMMA).
//   xv   = dropout(x_vals) / keep_prob
//   X    = densify(xv)                 [N x 256] f32   (1.6M atomics, L2-resident)
//   h    = X @ W  via v_wmma_f32_16x16x32_bf16         [N x 64] f32 (L2-resident)
//   agg  = adj_coo @ h  via f32 L2 atomics             -> d_out
//   out  = relu(agg)
// ---------------------------------------------------------------------------

#define N_NODES   100000
#define IN_DIM    256
#define OUT_DIM   64
#define KEEP_INV  (1.0f / 0.9f)

typedef __attribute__((ext_vector_type(16))) __bf16 v16bf;
typedef __attribute__((ext_vector_type(8)))  float  v8f;

// ---------------------------------------------------------------- zero fill
__global__ void zero_f32_kernel(float* __restrict__ p, int n4) {
    int i = blockIdx.x * blockDim.x + threadIdx.x;
    int stride = gridDim.x * blockDim.x;
    float4 z = make_float4(0.f, 0.f, 0.f, 0.f);
    for (; i < n4; i += stride) ((float4*)p)[i] = z;
}

// ---------------------------------------- scatter x (with dropout) -> dense X
__global__ void scatter_x_dense_kernel(const float* __restrict__ xv,
                                       const int* __restrict__ xr,
                                       const int* __restrict__ xc,
                                       const unsigned char* __restrict__ keep,
                                       float* __restrict__ X, int nnz) {
    int i = blockIdx.x * blockDim.x + threadIdx.x;
    if (i >= nnz) return;
    if (!keep[i]) return;
    float v = xv[i] * KEEP_INV;
    atomicAdd(&X[(size_t)xr[i] * IN_DIM + xc[i]], v);
}

// ------------------------- pre-pack W (f32 row-major) into WMMA B fragments
// Wfrag layout: [kb(8)][nw(4)][lane(32)][elem(16)] bf16, 32 KB total.
// B 32x16 bf16 lane map (ISA 7.12.2): lanes 0-15 -> N=lane, K=kb*32+{0..7,16..23};
//                                     lanes 16-31 -> N=lane-16, K=kb*32+{8..15,24..31}.
__global__ void pack_w_kernel(const float* __restrict__ W, __bf16* __restrict__ wf) {
    int tid = blockIdx.x * blockDim.x + threadIdx.x;   // 0..1023
    if (tid >= 8 * 4 * 32) return;
    int lane = tid & 31;
    int idx  = tid >> 5;
    int kb   = idx >> 2;
    int nw   = idx & 3;
    int half = lane >> 4;
    int n    = nw * 16 + (lane & 15);
    __bf16* dst = wf + (size_t)tid * 16;
    for (int i = 0; i < 16; ++i) {
        int k = kb * 32 + (i < 8 ? half * 8 + i : 16 + half * 8 + (i - 8));
        dst[i] = (__bf16)W[k * OUT_DIM + n];
    }
}

// ------------------------------------------------ dense GEMM h = X @ W (WMMA)
// Block: 256 threads = 8 waves. Block tile: 32 rows x 64 cols.
// Wave (mw,nw) -> one 16x16 WMMA tile; K = 256 in 8 steps of 32.
__global__ __launch_bounds__(256)
void gemm_wmma_kernel(const float* __restrict__ X,
                      const __bf16* __restrict__ wf,
                      float* __restrict__ h) {
    unsigned lane = threadIdx.x & 31u;
    unsigned wave = threadIdx.x >> 5;          // 0..7
    unsigned mw   = wave >> 2;                 // 0..1
    unsigned nw   = wave & 3;                  // 0..3
    unsigned half = lane >> 4;                 // 0 or 1
    int mbase = blockIdx.x * 32 + mw * 16;
    int row   = mbase + (lane & 15);

    v8f c = {};
    const float* xrow = X + (size_t)row * IN_DIM;

#pragma unroll
    for (int kb = 0; kb < 8; ++kb) {
        // --- A fragment: 16x32 bf16 (ISA 7.12.2 16-bit A layout) ---
        // half==0: K = kb*32 + {0..7, 16..23}; half==1: K = kb*32 + {8..15, 24..31}
        const float* p = xrow + kb * 32 + half * 8;
        float ta[16];
        *(float4*)&ta[0]  = *(const float4*)(p + 0);
        *(float4*)&ta[4]  = *(const float4*)(p + 4);
        *(float4*)&ta[8]  = *(const float4*)(p + 16);
        *(float4*)&ta[12] = *(const float4*)(p + 20);
        v16bf a;
#pragma unroll
        for (int i = 0; i < 16; ++i) a[i] = (__bf16)ta[i];

        // --- B fragment: pre-packed, one 32B load ---
        v16bf b = *(const v16bf*)(wf + ((size_t)(kb * 4 + nw) * 32 + lane) * 16);

        c = __builtin_amdgcn_wmma_f32_16x16x32_bf16(
                /*neg_a=*/false, a, /*neg_b=*/false, b,
                /*c_mod=*/(short)0, c, /*reuse_a=*/false, /*reuse_b=*/false);
    }

    // C/D layout: elem j, lanes 0-15 -> M=j, N=lane; lanes 16-31 -> M=j+8, N=lane-16
    int n = nw * 16 + (lane & 15);
#pragma unroll
    for (int j = 0; j < 8; ++j) {
        int m = mbase + j + (half ? 8 : 0);
        h[(size_t)m * OUT_DIM + n] = c[j];
    }
}

// ------------------------------- SpMM2: agg[row,:] += v * h[col,:]  (atomics)
// 16 lanes per nnz, float4 gather of h (L2-resident), 4 f32 atomics per lane.
__global__ void spmm2_scatter_kernel(const float* __restrict__ av,
                                     const int* __restrict__ ar,
                                     const int* __restrict__ ac,
                                     const float* __restrict__ h,
                                     float* __restrict__ out, int nnz) {
    long long tid = (long long)blockIdx.x * blockDim.x + threadIdx.x;
    long long e = tid >> 4;
    if (e >= nnz) return;
    int q = (int)(tid & 15);
    int r = ar[e], cidx = ac[e];
    float v = av[e];
    float4 hv = ((const float4*)h)[(size_t)cidx * 16 + q];
    float* dst = out + (size_t)r * OUT_DIM + q * 4;
    atomicAdd(dst + 0, v * hv.x);
    atomicAdd(dst + 1, v * hv.y);
    atomicAdd(dst + 2, v * hv.z);
    atomicAdd(dst + 3, v * hv.w);
}

// ------------------------ fallback SpMM1 (no dense X): h[row,:] += v * W[col,:]
__global__ void spmm1_scatter_kernel(const float* __restrict__ xv,
                                     const int* __restrict__ xr,
                                     const int* __restrict__ xc,
                                     const unsigned char* __restrict__ keep,
                                     const float* __restrict__ W,
                                     float* __restrict__ h, int nnz) {
    long long tid = (long long)blockIdx.x * blockDim.x + threadIdx.x;
    long long e = tid >> 4;
    if (e >= nnz) return;
    int q = (int)(tid & 15);
    if (!keep[e]) return;
    int r = xr[e], cidx = xc[e];
    float v = xv[e] * KEEP_INV;
    float4 wv = ((const float4*)W)[(size_t)cidx * 16 + q];
    float* dst = h + (size_t)r * OUT_DIM + q * 4;
    atomicAdd(dst + 0, v * wv.x);
    atomicAdd(dst + 1, v * wv.y);
    atomicAdd(dst + 2, v * wv.z);
    atomicAdd(dst + 3, v * wv.w);
}

// ------------------------------------------------------------------- ReLU
__global__ void relu_kernel(float* __restrict__ p, int n) {
    int i = blockIdx.x * blockDim.x + threadIdx.x;
    int stride = gridDim.x * blockDim.x;
    for (; i < n; i += stride) p[i] = fmaxf(p[i], 0.0f);
}

// ---------------------------------------------------------------------------
extern "C" void kernel_launch(void* const* d_in, const int* in_sizes, int n_in,
                              void* d_out, int out_size, void* d_ws, size_t ws_size,
                              hipStream_t stream) {
    const float*         x_vals   = (const float*)d_in[0];
    const int*           x_rows   = (const int*)d_in[1];
    const int*           x_cols   = (const int*)d_in[2];
    const float*         adj_vals = (const float*)d_in[3];
    const int*           adj_rows = (const int*)d_in[4];
    const int*           adj_cols = (const int*)d_in[5];
    const float*         W        = (const float*)d_in[6];
    const unsigned char* keep     = (const unsigned char*)d_in[7];

    const int x_nnz = in_sizes[0];
    const int a_nnz = in_sizes[3];
    float* out = (float*)d_out;

    const size_t X_bytes  = (size_t)N_NODES * IN_DIM * sizeof(float);   // 102.4 MB
    const size_t Wf_bytes = (size_t)8 * 4 * 32 * 16 * sizeof(__bf16);   // 32 KB
    const size_t h_bytes  = (size_t)N_NODES * OUT_DIM * sizeof(float);  // 25.6 MB
    const size_t need = X_bytes + 256 + Wf_bytes + 256 + h_bytes;

    // zero the output accumulator (harness poisons it)
    zero_f32_kernel<<<2048, 256, 0, stream>>>(out, out_size / 4);

    if (ws_size >= need) {
        // ---------------- main path: densify -> WMMA GEMM -> scatter ----------
        char* base = (char*)d_ws;
        float*  X  = (float*)base;
        __bf16* wf = (__bf16*)(base + ((X_bytes + 255) & ~(size_t)255));
        float*  h  = (float*)((char*)wf + ((Wf_bytes + 255) & ~(size_t)255));

        zero_f32_kernel<<<4096, 256, 0, stream>>>(X, (int)(X_bytes / 16));
        pack_w_kernel<<<4, 256, 0, stream>>>(W, wf);
        scatter_x_dense_kernel<<<(x_nnz + 255) / 256, 256, 0, stream>>>(
            x_vals, x_rows, x_cols, keep, X, x_nnz);
        gemm_wmma_kernel<<<N_NODES / 32, 256, 0, stream>>>(X, wf, h);

        long long t2 = (long long)a_nnz * 16;
        spmm2_scatter_kernel<<<(unsigned)((t2 + 255) / 256), 256, 0, stream>>>(
            adj_vals, adj_rows, adj_cols, h, out, a_nnz);
    } else if (ws_size >= h_bytes) {
        // ---------------- fallback: all-atomic path (no dense X) --------------
        float* h = (float*)d_ws;
        zero_f32_kernel<<<2048, 256, 0, stream>>>(h, (int)(h_bytes / 16));
        long long t1 = (long long)x_nnz * 16;
        spmm1_scatter_kernel<<<(unsigned)((t1 + 255) / 256), 256, 0, stream>>>(
            x_vals, x_rows, x_cols, keep, W, h, x_nnz);
        long long t2 = (long long)a_nnz * 16;
        spmm2_scatter_kernel<<<(unsigned)((t2 + 255) / 256), 256, 0, stream>>>(
            adj_vals, adj_rows, adj_cols, h, out, a_nnz);
    }

    relu_kernel<<<2048, 256, 0, stream>>>(out, out_size);
}
